// Linear2bit_73315091743455
// MI455X (gfx1250) — matrix operand order repaired and checked by reading, not compiled, not run on hardware
//
#include <hip/hip_runtime.h>
#include <hip/hip_bf16.h>

typedef __attribute__((ext_vector_type(16))) _Float16 v16h;
typedef __attribute__((ext_vector_type(8)))  _Float16 v8h;
typedef __attribute__((ext_vector_type(8)))  float    v8f;

// Problem constants (match reference)
#define IN_F   4096
#define OUT_F  4096
#define NGRP   32      // IN_F / 128
#define BM     128
#define BN     128
#define BK     64
#define LDT    72      // padded LDS stride in halves (144 B, 16B-aligned rows)

__global__ __launch_bounds__(256)
void Linear2bit_wmma_kernel(const float* __restrict__ x,
                            const int*   __restrict__ wp,
                            const float* __restrict__ scales,
                            const float* __restrict__ zeros,
                            const float* __restrict__ bias,
                            float* __restrict__ out)
{
    __shared__ _Float16 As[BM * LDT];   // x tile, [m][k], f16
    __shared__ _Float16 Bs[BN * LDT];   // dequantized weight tile, [n][k], f16

    const int tid  = threadIdx.x;
    const int lane = tid & 31;
    const int wave = tid >> 5;
    const int lr   = lane & 15;    // lane row within half-wave
    const int hi   = lane >> 4;    // half-wave select

    const int waveM = wave & 3;    // 4 waves along M (32 rows each)
    const int waveN = wave >> 2;   // 2 waves along N (64 cols each)

    const int bM = blockIdx.y * BM;
    const int bN = blockIdx.x * BN;

    // staging assignment: each thread owns one (row, 32-wide k-chunk)
    const int srow = tid >> 1;           // 0..127
    const int scb  = (tid & 1) * 32;     // 0 or 32

    const float* xg = x  + (size_t)(bM + srow) * IN_F + scb;
    const int*   wg = wp + (size_t)(bN + srow) * (IN_F / 4) + (scb >> 2);
    const float* sg = scales + (size_t)(bN + srow) * NGRP;
    const float* zg = zeros  + (size_t)(bN + srow) * NGRP;

    v8f acc[2][4] = {};

    // --- register pipeline: tile staged in registers one iteration ahead ---
    float4 xr[8];
    int4   wr[2];

    // prologue: load tile k0=0
    {
        const float4* xa = (const float4*)(xg);
        #pragma unroll
        for (int j = 0; j < 8; ++j) xr[j] = xa[j];
        const int4* wa = (const int4*)(wg);
        wr[0] = wa[0];
        wr[1] = wa[1];
    }

    for (int k0 = 0; k0 < IN_F; k0 += BK) {
        // ---------------- drain regs: convert + store tile k0 into LDS ----
        {
            _Float16* adst = As + srow * LDT + scb;
            #pragma unroll
            for (int j = 0; j < 4; ++j) {
                float4 f0 = xr[2 * j];
                float4 f1 = xr[2 * j + 1];
                v8h h;
                h[0] = (_Float16)f0.x; h[1] = (_Float16)f0.y;
                h[2] = (_Float16)f0.z; h[3] = (_Float16)f0.w;
                h[4] = (_Float16)f1.x; h[5] = (_Float16)f1.y;
                h[6] = (_Float16)f1.z; h[7] = (_Float16)f1.w;
                *(v8h*)(adst + 8 * j) = h;
            }
            const int g = (k0 + scb) >> 7;     // group of 128 along K
            const float s = sg[g];
            const float z = zg[g];
            int packs[8] = { wr[0].x, wr[0].y, wr[0].z, wr[0].w,
                             wr[1].x, wr[1].y, wr[1].z, wr[1].w };
            _Float16* bdst = Bs + srow * LDT + scb;
            #pragma unroll
            for (int j = 0; j < 8; j += 2) {
                v8h h;
                #pragma unroll
                for (int q = 0; q < 2; ++q) {
                    const int byte = packs[j + q];
                    #pragma unroll
                    for (int e = 0; e < 4; ++e) {
                        const int qv = (byte >> (2 * e)) & 3;
                        h[q * 4 + e] = (_Float16)((float)qv * s + z);
                    }
                }
                *(v8h*)(bdst + 4 * j) = h;
            }
        }
        __syncthreads();   // tile k0 visible in LDS

        // ------- issue globals for tile k0+BK; wait sinks past compute ----
        if (k0 + BK < IN_F) {
            const float4* xa = (const float4*)(xg + k0 + BK);
            #pragma unroll
            for (int j = 0; j < 8; ++j) xr[j] = xa[j];
            const int4* wa = (const int4*)(wg + ((k0 + BK) >> 2));
            wr[0] = wa[0];
            wr[1] = wa[1];
        }

        // ---------------- compute: 2 k-steps of 16x16x32 WMMA -------------
        #pragma unroll
        for (int ks = 0; ks < 2; ++ks) {
            // 16-bit A 16x32 layout: lanes 0-15 hold K {0-7,16-23}, lanes
            // 16-31 hold K {8-15,24-31} (two contiguous 8-half groups).
            union frag { v16h v; v8h h[2]; };
            frag a0, a1, b[4];
            const int kA = ks * 32 + hi * 8;
            const _Float16* ap0 = As + (waveM * 32 + lr) * LDT;
            const _Float16* ap1 = ap0 + 16 * LDT;
            a0.h[0] = *(const v8h*)(ap0 + kA);
            a0.h[1] = *(const v8h*)(ap0 + kA + 16);
            a1.h[0] = *(const v8h*)(ap1 + kA);
            a1.h[1] = *(const v8h*)(ap1 + kA + 16);

            // B 32x16 layout: same K order as A: lanes 0-15 hold K {0-7,16-23},
            // lanes 16-31 hold K {8-15,24-31} (two 8-half groups in [n][k]).
            #pragma unroll
            for (int tn = 0; tn < 4; ++tn) {
                const _Float16* bp =
                    Bs + (waveN * 64 + tn * 16 + lr) * LDT + ks * 32 + hi * 8;
                b[tn].h[0] = *(const v8h*)(bp);
                b[tn].h[1] = *(const v8h*)(bp + 16);
            }

            #pragma unroll
            for (int tn = 0; tn < 4; ++tn) {
                acc[0][tn] = __builtin_amdgcn_wmma_f32_16x16x32_f16(
                    false, a0.v, false, b[tn].v, (short)0, acc[0][tn],
                    false, false);
                acc[1][tn] = __builtin_amdgcn_wmma_f32_16x16x32_f16(
                    false, a1.v, false, b[tn].v, (short)0, acc[1][tn],
                    false, false);
            }
        }
        __syncthreads();   // all LDS reads done before next overwrite
    }

    // ---------------- epilogue: bias add + fp32 NT store ------------------
    // C/D layout: VGPR r -> M = r (lanes 0-15) / r+8 (lanes 16-31), N = lane&15.
    #pragma unroll
    for (int tm = 0; tm < 2; ++tm) {
        const int gm = bM + waveM * 32 + tm * 16 + hi * 8;
        #pragma unroll
        for (int tn = 0; tn < 4; ++tn) {
            const int gn = bN + waveN * 64 + tn * 16 + lr;
            const float bv = bias[gn];
            #pragma unroll
            for (int r = 0; r < 8; ++r) {
                __builtin_nontemporal_store(acc[tm][tn][r] + bv,
                                            out + (size_t)(gm + r) * OUT_F + gn);
            }
        }
    }
}

extern "C" void kernel_launch(void* const* d_in, const int* in_sizes, int n_in,
                              void* d_out, int out_size, void* d_ws, size_t ws_size,
                              hipStream_t stream) {
    (void)n_in; (void)d_ws; (void)ws_size; (void)out_size;
    const float* x      = (const float*)d_in[0];
    const int*   wp     = (const int*)  d_in[1];
    const float* scales = (const float*)d_in[2];
    const float* zeros  = (const float*)d_in[3];
    const float* bias   = (const float*)d_in[4];
    float* out = (float*)d_out;

    const int M = in_sizes[0] / IN_F;   // 8192 tokens
    dim3 grid(OUT_F / BN, M / BM);      // 32 x 64 blocks
    dim3 block(256);
    Linear2bit_wmma_kernel<<<grid, block, 0, stream>>>(x, wp, scales, zeros, bias, out);
}
